// CAUMUserEncoder_82884278878390
// MI455X (gfx1250) — matrix operand order repaired and checked by reading, not compiled, not run on hardware
//
#include <hip/hip_runtime.h>

// Problem constants (reference: D=400, H=20 heads, B=64, NC=10, NH=50)
#define DD    400
#define NH_   50
#define NC_   10
#define NB_   64
#define NHEAD 20
#define DH    20

typedef __bf16 v16bf __attribute__((ext_vector_type(16)));
typedef float  v8f   __attribute__((ext_vector_type(8)));

union Frag { v16bf v; unsigned short u[16]; };

__device__ __forceinline__ unsigned short f2bfu(float f) {
  unsigned u = __builtin_bit_cast(unsigned, f);
  unsigned r = u + 0x7FFFu + ((u >> 16) & 1u);   // round-to-nearest-even bf16
  return (unsigned short)(r >> 16);
}
__device__ __forceinline__ float bfu2f(unsigned short s) {
  unsigned u = ((unsigned)s) << 16;
  return __builtin_bit_cast(float, u);
}

// ---------------------------------------------------------------------------
// Tiled WMMA GEMM:  C = act( A @ W^T + bias + epilogue )   (bf16 MACs, f32 acc)
// AMODE: 0 = A f32 row-major (lda) ; 1 = A bf16(ushort) row-major ;
//        2 = ROLL3 gather over h:  row->(b,i), K=1200, segs (i-1, i, i+1) mod NH
// BT   : 0 -> B[n,k]=Bw[n*ldb+k] ; 1 -> B[n,k]=Bw[k*ldb+n]   (transposed weight read)
// EPI  : 0 none
//        1 += add0[(row/NH_)*N + col]
//        2 += add0[((row/(NH_*NC_))*NH_ + row%NH_)*N + col] + add1[(row/NH_)*N + col]
// ACT  : 0 none ; 1 tanh
// OUTBF: 0 f32 out ; 1 bf16(ushort) out
// Block: 256 threads = 8 wave32, 64x64 output tile, BK=32.
// ---------------------------------------------------------------------------
template<int AMODE, int BT, int EPI, int ACT, int OUTBF>
__global__ __launch_bounds__(256) void gemm_k(const void* __restrict__ Aptr, int lda,
                                              const float* __restrict__ Bw, int ldb,
                                              const float* __restrict__ bias,
                                              const float* __restrict__ add0,
                                              const float* __restrict__ add1,
                                              void* __restrict__ Cout,
                                              int M, int N, int K)
{
  __shared__ __align__(16) unsigned short As[64][40];  // rows x k (pad 32->40 halves)
  __shared__ __align__(16) unsigned short Bs[64][40];  // cols(n) x k
  const int tid  = threadIdx.x;
  const int lane = tid & 31;
  const int wid  = tid >> 5;
  const int wm   = wid & 3;           // wave row group: 16 rows each
  const int wn   = wid >> 2;          // wave col group: 32 cols each
  const int g    = lane >> 4;
  const int bm   = blockIdx.y * 64;
  const int bn   = blockIdx.x * 64;

  v8f acc0 = {0.f,0.f,0.f,0.f,0.f,0.f,0.f,0.f};
  v8f acc1 = acc0;

  const int r0 = tid >> 2;            // 0..63
  const int c0 = (tid & 3) * 8;       // 0,8,16,24

  for (int k0 = 0; k0 < K; k0 += 32) {
    // ---- stage A tile (64x32) into LDS as bf16 ----
    {
      const int gr = bm + r0;
      #pragma unroll
      for (int j = 0; j < 8; ++j) {
        const int k = k0 + c0 + j;
        float v = 0.f;
        if (k < K) {
          if (AMODE == 0) {
            if (gr < M) v = ((const float*)Aptr)[(long)gr * lda + k];
          } else if (AMODE == 1) {
            if (gr < M) v = bfu2f(((const unsigned short*)Aptr)[(long)gr * lda + k]);
          } else {                      // ROLL3 over h
            if (gr < M) {
              const int b = gr / NH_, i = gr % NH_;
              const int seg = k / DD, kk = k - seg * DD;
              const int si = (seg == 0) ? (i + NH_ - 1) % NH_
                           : (seg == 2) ? (i + 1) % NH_ : i;
              v = ((const float*)Aptr)[((long)(b * NH_ + si)) * DD + kk];
            }
          }
        }
        As[r0][c0 + j] = f2bfu(v);
      }
      // ---- stage B tile: Bs[n_local][k_local] = W[n][k] ----
      const int gn = bn + r0;
      #pragma unroll
      for (int j = 0; j < 8; ++j) {
        const int k = k0 + c0 + j;
        float v = 0.f;
        if (k < K && gn < N) v = BT ? Bw[(long)k * ldb + gn] : Bw[(long)gn * ldb + k];
        Bs[r0][c0 + j] = f2bfu(v);
      }
    }
    __syncthreads();

    // ---- WMMA: A frag (16x32) + two B frags (32x16) per wave ----
    Frag a;
    {
      const int row = wm * 16 + (lane & 15);
      __builtin_memcpy(&a.u[0], &As[row][16 * g], 32);
    }
    {
      Frag b;
      const int col0 = wn * 32 + (lane & 15);
      __builtin_memcpy(&b.u[0], &Bs[col0][16 * g], 32);
      acc0 = __builtin_amdgcn_wmma_f32_16x16x32_bf16(false, a.v, false, b.v,
                                                     (short)0, acc0, false, false);
      __builtin_memcpy(&b.u[0], &Bs[col0 + 16][16 * g], 32);
      acc1 = __builtin_amdgcn_wmma_f32_16x16x32_bf16(false, a.v, false, b.v,
                                                     (short)0, acc1, false, false);
    }
    __syncthreads();
  }

  // ---- epilogue (C layout: lane holds col=L%16, rows i+8*(L/16)) ----
  #pragma unroll
  for (int s = 0; s < 2; ++s) {
    const v8f acc = s ? acc1 : acc0;
    const int col = bn + wn * 32 + s * 16 + (lane & 15);
    if (col >= N) continue;
    const float bb = bias ? bias[col] : 0.f;
    #pragma unroll
    for (int i = 0; i < 8; ++i) {
      const int row = bm + wm * 16 + i + 8 * g;
      if (row >= M) continue;
      float v = acc[i] + bb;
      if (EPI == 1) {
        v += add0[(long)(row / NH_) * N + col];
      } else if (EPI == 2) {
        const int b  = row / (NH_ * NC_);
        const int nh = row % NH_;
        v += add0[(long)(b * NH_ + nh) * N + col] + add1[(long)(row / NH_) * N + col];
      }
      if (ACT) v = tanhf(v);
      if (OUTBF) ((unsigned short*)Cout)[(long)row * N + col] = f2bfu(v);
      else       ((float*)Cout)[(long)row * N + col] = v;
    }
  }
}

// ---------------------------------------------------------------------------
// Flash attention over qkv = Rq[(b,nh)] + Sq[(b,nc)] + bqkv (factored QKV).
// One block per ((nh,head), 64-row l-strip); 4 wave32, 16 l-rows per wave.
// dh=20 padded to 32; scores/P/PV all via v_wmma_f32_16x16x32_bf16.
// Output O[(l*NH+nh)*400 + head*20 + d] in bf16 for the fused Wc GEMM.
// ---------------------------------------------------------------------------
__global__ __launch_bounds__(128) void attn_k(const float* __restrict__ Rq,
                                              const float* __restrict__ Sq,
                                              const float* __restrict__ bqkv,
                                              unsigned short* __restrict__ O)
{
  const int pair = blockIdx.x;
  const int n    = pair / NHEAD;       // MHA batch index == nh
  const int head = pair % NHEAD;
  const int l0   = blockIdx.y * 64;
  const int tid  = threadIdx.x;
  const int lane = tid & 31;
  const int wid  = tid >> 5;
  const int g    = lane >> 4;

  __shared__ __align__(16) unsigned short Qs[64][40];      // l x dh(pad32)
  __shared__ __align__(16) unsigned short Ks[64][40];      // m x dh
  __shared__ __align__(16) unsigned short Vt[32][72];      // dh x m(pad)
  __shared__ __align__(16) unsigned short Ps[4][16][72];   // per-wave P: l x m

  const float qscale = rsqrtf((float)DH);
  const int   e0     = head * DH;

  // build Q strip once (1/sqrt(dh) folded in)
  for (int idx = tid; idx < 64 * 32; idx += 128) {
    const int r = idx >> 5, d = idx & 31;
    float v = 0.f;
    if (d < DH) {
      const int l = l0 + r;
      const int b = l / NC_;
      const int e = e0 + d;
      v = (Rq[(long)(b * NH_ + n) * 1200 + e] + Sq[(long)l * 1200 + e] + bqkv[e]) * qscale;
    }
    Qs[r][d] = f2bfu(v);
  }
  __syncthreads();

  float run_m[8], run_s[8];
  #pragma unroll
  for (int i = 0; i < 8; ++i) { run_m[i] = -1e30f; run_s[i] = 0.f; }
  v8f o0 = {0.f,0.f,0.f,0.f,0.f,0.f,0.f,0.f};
  v8f o1 = o0;

  Frag qa;                              // loop-invariant Q fragment
  {
    const int row = wid * 16 + (lane & 15);
    __builtin_memcpy(&qa.u[0], &Qs[row][16 * g], 32);
  }

  for (int mt = 0; mt < 10; ++mt) {
    const int m0 = mt * 64;
    __syncthreads();                    // previous K/V reads done
    for (int idx = tid; idx < 64 * 32; idx += 128) {
      const int r = idx >> 5, d = idx & 31;
      float kv = 0.f, vv = 0.f;
      if (d < DH) {
        const int m = m0 + r;
        const int b = m / NC_;
        const int e = e0 + d;
        const long rb = (long)(b * NH_ + n) * 1200;
        const long rm = (long)m * 1200;
        kv = Rq[rb + 400 + e] + Sq[rm + 400 + e] + bqkv[400 + e];
        vv = Rq[rb + 800 + e] + Sq[rm + 800 + e] + bqkv[800 + e];
      }
      Ks[r][d] = f2bfu(kv);
      Vt[d][r] = f2bfu(vv);
    }
    __syncthreads();

    // scores S = Q K^T : 4 x (16x16) tiles per wave
    v8f st[4];
    #pragma unroll
    for (int t = 0; t < 4; ++t) {
      Frag kb;
      const int col = t * 16 + (lane & 15);
      __builtin_memcpy(&kb.u[0], &Ks[col][16 * g], 32);
      v8f z = {0.f,0.f,0.f,0.f,0.f,0.f,0.f,0.f};
      st[t] = __builtin_amdgcn_wmma_f32_16x16x32_bf16(false, qa.v, false, kb.v,
                                                      (short)0, z, false, false);
    }

    // online softmax update (row reductions across the 16-lane half)
    float nm[8], corr[8], rs[8];
    #pragma unroll
    for (int i = 0; i < 8; ++i) {
      float mx = fmaxf(fmaxf(st[0][i], st[1][i]), fmaxf(st[2][i], st[3][i]));
      for (int d = 1; d < 16; d <<= 1) mx = fmaxf(mx, __shfl_xor(mx, d, 32));
      nm[i]   = fmaxf(run_m[i], mx);
      corr[i] = __expf(run_m[i] - nm[i]);
      rs[i]   = 0.f;
    }
    #pragma unroll
    for (int t = 0; t < 4; ++t) {
      const int col = t * 16 + (lane & 15);
      #pragma unroll
      for (int i = 0; i < 8; ++i) {
        const float p = __expf(st[t][i] - nm[i]);
        rs[i] += p;
        Ps[wid][i + 8 * g][col] = f2bfu(p);
      }
    }
    #pragma unroll
    for (int i = 0; i < 8; ++i) {
      float s = rs[i];
      for (int d = 1; d < 16; d <<= 1) s += __shfl_xor(s, d, 32);
      run_s[i] = run_s[i] * corr[i] + s;
      run_m[i] = nm[i];
      o0[i] *= corr[i];
      o1[i] *= corr[i];
    }
    // cross-lane LDS RAW inside the wave: drain DS stores before P reads
    asm volatile("s_wait_dscnt 0" ::: "memory");

    // O += P V : P is 16x64 A-operand, two K-slices of 32
    #pragma unroll
    for (int s2 = 0; s2 < 2; ++s2) {
      Frag pa;
      const int row = lane & 15;
      __builtin_memcpy(&pa.u[0], &Ps[wid][row][s2 * 32 + 16 * g], 32);
      Frag vb;
      const int col0 = lane & 15;
      __builtin_memcpy(&vb.u[0], &Vt[col0][s2 * 32 + 16 * g], 32);
      o0 = __builtin_amdgcn_wmma_f32_16x16x32_bf16(false, pa.v, false, vb.v,
                                                   (short)0, o0, false, false);
      __builtin_memcpy(&vb.u[0], &Vt[col0 + 16][s2 * 32 + 16 * g], 32);
      o1 = __builtin_amdgcn_wmma_f32_16x16x32_bf16(false, pa.v, false, vb.v,
                                                   (short)0, o1, false, false);
    }
  }

  // epilogue: O[l, nh, head*20+d] = acc/rowsum  (bf16)
  #pragma unroll
  for (int ns = 0; ns < 2; ++ns) {
    const int d = ns * 16 + (lane & 15);
    if (d >= DH) continue;
    #pragma unroll
    for (int i = 0; i < 8; ++i) {
      const int l = l0 + wid * 16 + i + 8 * g;
      const float val = (ns ? o1[i] : o0[i]) / run_s[i];
      const long  r = (long)l * NH_ + n;
      O[r * DD + e0 + d] = f2bfu(val);
    }
  }
}

// cvec[n] = b3[n] + sum_j W3b[n,j]*bo[j]    (W3b = W3[:,400:800])
__global__ void cvec_k(const float* __restrict__ W3, const float* __restrict__ b3,
                       const float* __restrict__ bo, float* __restrict__ cvec)
{
  const int n = blockIdx.x * blockDim.x + threadIdx.x;
  if (n < DD) {
    float s = b3[n];
    for (int j = 0; j < DD; ++j) s += W3[(long)n * 800 + 400 + j] * bo[j];
    cvec[n] = s;
  }
}

// scores[r] = ba3 + a2[r,:200] . Wa3
__global__ void score_k(const unsigned short* __restrict__ a2,
                        const float* __restrict__ Wa3, const float* __restrict__ ba3,
                        float* __restrict__ scores)
{
  const int r = blockIdx.x * blockDim.x + threadIdx.x;
  if (r < NB_ * NC_ * NH_) {
    float s = ba3[0];
    const unsigned short* p = a2 + (long)r * 200;
    for (int j = 0; j < 200; ++j) s += bfu2f(p[j]) * Wa3[j];
    scores[r] = s;
  }
}

// softmax over nh + weighted sum of h_all -> u[b,nc,:]
__global__ __launch_bounds__(256) void final_k(const float* __restrict__ scores,
                                               const float* __restrict__ hall,
                                               float* __restrict__ out)
{
  const int l   = blockIdx.x;             // 0..639  (b*NC+nc)
  const int tid = threadIdx.x;
  __shared__ float w[NH_];
  __shared__ float sm;
  if (tid < NH_) w[tid] = scores[(long)l * NH_ + tid];
  __syncthreads();
  if (tid == 0) {
    float m = w[0];
    for (int i = 1; i < NH_; ++i) m = fmaxf(m, w[i]);
    float s = 0.f;
    for (int i = 0; i < NH_; ++i) { w[i] = __expf(w[i] - m); s += w[i]; }
    sm = s;
  }
  __syncthreads();
  const float inv = 1.f / sm;
  for (int d = tid; d < DD; d += 256) {
    float acc = 0.f;
    for (int i = 0; i < NH_; ++i) acc += w[i] * hall[((long)l * NH_ + i) * DD + d];
    out[(long)l * DD + d] = acc * inv;
  }
}

extern "C" void kernel_launch(void* const* d_in, const int* in_sizes, int n_in,
                              void* d_out, int out_size, void* d_ws, size_t ws_size,
                              hipStream_t stream)
{
  (void)in_sizes; (void)n_in; (void)out_size; (void)ws_size;
  const float* h    = (const float*)d_in[0];
  const float* c    = (const float*)d_in[2];
  const float* W1   = (const float*)d_in[4];
  const float* b1   = (const float*)d_in[5];
  const float* W2   = (const float*)d_in[6];
  const float* b2   = (const float*)d_in[7];
  const float* W3   = (const float*)d_in[8];
  const float* b3   = (const float*)d_in[9];
  const float* Wa1  = (const float*)d_in[10];
  const float* ba1  = (const float*)d_in[11];
  const float* Wa2  = (const float*)d_in[12];
  const float* ba2  = (const float*)d_in[13];
  const float* Wa3  = (const float*)d_in[14];
  const float* ba3  = (const float*)d_in[15];
  const float* Wqkv = (const float*)d_in[16];
  const float* bqkv = (const float*)d_in[17];
  const float* Wo   = (const float*)d_in[18];
  const float* bo   = (const float*)d_in[19];

  char* ws = (char*)d_ws;
  size_t off = 0;
  auto take = [&](size_t bytes) {
    char* p = ws + off;
    off = (off + bytes + 255) & ~(size_t)255;
    return p;
  };
  float*          P    = (float*)take(3200UL * 400 * 4);
  float*          Q    = (float*)take(640UL * 400 * 4);
  float*          R    = (float*)take(3200UL * 400 * 4);
  float*          S    = (float*)take(640UL * 400 * 4);
  float*          Rq   = (float*)take(3200UL * 1200 * 4);
  float*          Sq   = (float*)take(640UL * 1200 * 4);
  float*          Pw3  = (float*)take(3200UL * 400 * 4);
  float*          Qw3  = (float*)take(640UL * 400 * 4);
  float*          Wc   = (float*)take(400UL * 400 * 4);
  float*          cvec = (float*)take(400UL * 4);
  float*          Cw   = (float*)take(640UL * 400 * 4);
  unsigned short* Obuf = (unsigned short*)take(32000UL * 400 * 2);
  float*          hall = (float*)take(32000UL * 400 * 4);
  unsigned short* a1   = (unsigned short*)take(32000UL * 400 * 2);
  unsigned short* a2   = (unsigned short*)take(32000UL * 200 * 2);
  float*          sco  = (float*)take(32000UL * 4);

  const dim3 blk(256);
  auto grid = [](int N, int M) { return dim3((N + 63) / 64, (M + 63) / 64); };

  // P(b,nh)  = h_{i-1}@W1a^T + h_i@W1b^T + h_{i+1}@W1c^T          (ROLL3, K=1200)
  gemm_k<2,0,0,0,0><<<grid(400,3200), blk, 0, stream>>>(h, 0, W1, 1600, nullptr, nullptr, nullptr, P, 3200, 400, 1200);
  // Q(b,nc)  = c@W1d^T + b1
  gemm_k<0,0,0,0,0><<<grid(400,640),  blk, 0, stream>>>(c, 400, W1 + 1200, 1600, b1, nullptr, nullptr, Q, 640, 400, 400);
  // R(b,nh)  = h@W2b^T ; S(b,nc) = c@W2a^T + b2      (h_sa = R + S)
  gemm_k<0,0,0,0,0><<<grid(400,3200), blk, 0, stream>>>(h, 400, W2 + 400, 800, nullptr, nullptr, nullptr, R, 3200, 400, 400);
  gemm_k<0,0,0,0,0><<<grid(400,640),  blk, 0, stream>>>(c, 400, W2, 800, b2, nullptr, nullptr, S, 640, 400, 400);
  // factored QKV projection
  gemm_k<0,0,0,0,0><<<grid(1200,3200), blk, 0, stream>>>(R, 400, Wqkv, 400, nullptr, nullptr, nullptr, Rq, 3200, 1200, 400);
  gemm_k<0,0,0,0,0><<<grid(1200,640),  blk, 0, stream>>>(S, 400, Wqkv, 400, bqkv, nullptr, nullptr, Sq, 640, 1200, 400);
  // h_cnn @ W3a^T split: Pw3 + Qw3
  gemm_k<0,0,0,0,0><<<grid(400,3200), blk, 0, stream>>>(P, 400, W3, 800, nullptr, nullptr, nullptr, Pw3, 3200, 400, 400);
  gemm_k<0,0,0,0,0><<<grid(400,640),  blk, 0, stream>>>(Q, 400, W3, 800, nullptr, nullptr, nullptr, Qw3, 640, 400, 400);
  // Wc = W3b @ Wo  (B read transposed) ; cvec = b3 + W3b@bo
  gemm_k<0,1,0,0,0><<<grid(400,400),  blk, 0, stream>>>(W3 + 400, 800, Wo, 400, nullptr, nullptr, nullptr, Wc, 400, 400, 400);
  cvec_k<<<2, 256, 0, stream>>>(W3, b3, bo, cvec);
  // Cw(b,nc) = c@Wa1b^T + ba1
  gemm_k<0,0,0,0,0><<<grid(400,640),  blk, 0, stream>>>(c, 400, Wa1 + 400, 800, ba1, nullptr, nullptr, Cw, 640, 400, 400);
  // flash attention -> O (bf16)
  attn_k<<<dim3(NH_ * NHEAD, 10), dim3(128), 0, stream>>>(Rq, Sq, bqkv, Obuf);
  // h_all = O@Wc^T + Pw3[(b,nh)] + Qw3[(b,nc)] + cvec
  gemm_k<1,0,2,0,0><<<grid(400,32000), blk, 0, stream>>>(Obuf, 400, Wc, 400, cvec, Pw3, Qw3, hall, 32000, 400, 400);
  // a1 = tanh(h_all@Wa1a^T + Cw[(b,nc)])              (bf16 out)
  gemm_k<0,0,1,1,1><<<grid(400,32000), blk, 0, stream>>>(hall, 400, Wa1, 800, nullptr, Cw, nullptr, a1, 32000, 400, 400);
  // a2 = tanh(a1@Wa2^T + ba2)                         (bf16 in+out)
  gemm_k<1,0,0,1,1><<<grid(200,32000), blk, 0, stream>>>(a1, 400, Wa2, 400, ba2, nullptr, nullptr, a2, 32000, 200, 400);
  // a3 logits, softmax over nh, weighted sum -> u
  score_k<<<(32000 + 255) / 256, 256, 0, stream>>>(a2, Wa3, ba3, sco);
  final_k<<<640, 256, 0, stream>>>(sco, hall, (float*)d_out);
}
